// MoECustomBasicBlock_64347200028677
// MI455X (gfx1250) — matrix operand order repaired and checked by reading, not compiled, not run on hardware
//
#include <hip/hip_runtime.h>
#include <math.h>

typedef __attribute__((ext_vector_type(16))) _Float16 v16h;
typedef __attribute__((ext_vector_type(8)))  float    v8f;

#define NOUT 18874368   // 8*128*96*192
#define LPAD 8

// ---------------------------------------------------------------------------
// A-matrix fragment packing (contiguous k order chosen here, B fragments match):
// lane l (0..31): M = l&15 ; kb = (l>=16)?16:0
// half j (0..15): K_local = kb + j
// Packed layout: [kstep][mtile][lane][16 halfs]  -> 32B contiguous per lane.
// ---------------------------------------------------------------------------
__device__ __forceinline__ int frag_index(int M, int K) {
  int kstep = K >> 5;
  int klo   = K & 31;
  int lane  = (M & 15) | ((klo >> 4) << 4);
  int j     = klo & 15;
  int mtile = M >> 4;
  return ((kstep * 8 + mtile) * 32 + lane) * 16 + j;
}

// B-matrix 32x16 fragment (contiguous k order, same as A packing): lane l: N = l&15,
// K_local = ((l>=16)?16:0) + j  -> 16 contiguous channels in our LDS layout.
__device__ __forceinline__ v16h load_bfrag(const _Float16* p) {
  v16h bf;
  *(float4*)&bf         = *(const float4*)(p);
  *(((float4*)&bf) + 1) = *(const float4*)(p + 8);
  return bf;
}

// ---------------------------------------------------------------------------
// Gates: logits = emb @ W + b ; top-32 sparse softmax (rank-count tie-break
// matches lax.top_k: lower index wins ties).
// ---------------------------------------------------------------------------
__global__ void gates_kernel(const float* __restrict__ emb,
                             const float* __restrict__ w1, const float* __restrict__ b1,
                             const float* __restrict__ w2, const float* __restrict__ b2,
                             float* __restrict__ gout) {
  int b = blockIdx.x;   // 0..7
  int g = blockIdx.y;   // 0..1
  int c = threadIdx.x;  // 0..127
  const float* w    = g ? w2 : w1;
  const float* bias = g ? b2 : b1;
  __shared__ float logits[128];
  __shared__ float red[128];
  float acc = bias[c];
  for (int e = 0; e < 128; ++e) acc += emb[b * 128 + e] * w[e * 128 + c];
  logits[c] = acc;
  __syncthreads();
  int rank = 0;
  float vmax = -3.4e38f;
  for (int j = 0; j < 128; ++j) {
    float vj = logits[j];
    rank += (vj > acc) || (vj == acc && j < c);
    vmax = fmaxf(vmax, vj);
  }
  bool keep = rank < 32;               // top-32
  float ex = keep ? __expf(acc - vmax) : 0.0f;
  red[c] = ex;
  __syncthreads();
  for (int s = 64; s > 0; s >>= 1) {
    if (c < s) red[c] += red[c + s];
    __syncthreads();
  }
  float gate = keep ? ex / red[0] : 0.0f;
  gout[g * 1024 + b * 128 + c] = gate;
}

// ---------------------------------------------------------------------------
// Weight packing: K order = (kh*3+kw)*Cin + ci  (channels contiguous per tap)
// ---------------------------------------------------------------------------
__global__ void pack_conv3x3(const float* __restrict__ w, _Float16* __restrict__ dst,
                             int Cin, int total) {
  int idx = blockIdx.x * 256 + threadIdx.x;
  if (idx >= total) return;
  int kw = idx % 3; int t = idx / 3;
  int kh = t % 3;   t /= 3;
  int ci = t % Cin; int o = t / Cin;
  int K = (kh * 3 + kw) * Cin + ci;
  dst[frag_index(o, K)] = (_Float16)w[idx];
}

__global__ void pack_ds(const float* __restrict__ w, const float* __restrict__ gamma,
                        _Float16* __restrict__ dst) {
  int idx = blockIdx.x * 256 + threadIdx.x;  // 128*64
  int ci = idx & 63; int o = idx >> 6;
  float scale = gamma[o] * rsqrtf(1.0f + 1e-5f);   // BN eval, var=1
  dst[frag_index(o, ci)] = (_Float16)(w[idx] * scale);
}

// ---------------------------------------------------------------------------
// conv1: 3x3 stride2 pad1, Cin=64 -> Cout=128 ; * g1 ; leaky ; store f16
// Block: b,h fixed, 64 output columns. 8 waves: wave = m-tile, 4 n-tiles each.
// ---------------------------------------------------------------------------
#define C1_COLS 130
#define C1_CIP  (64 + LPAD)    // 72

__global__ __launch_bounds__(256) void conv1_kernel(
    const float* __restrict__ x, const _Float16* __restrict__ packedA,
    const float* __restrict__ g1, _Float16* __restrict__ out2) {
  __shared__ _Float16 tile[3 * C1_COLS * C1_CIP];
  int w0 = blockIdx.x * 64;
  int h  = blockIdx.y;
  int b  = blockIdx.z;
  int tid = threadIdx.x;

  // Stage input halo [r=3][col=130][ci=64] (ci innermost, 16B-aligned runs).
  for (int idx = tid; idx < 3 * 64 * C1_COLS; idx += 256) {
    int col = idx % C1_COLS; int t = idx / C1_COLS;
    int ci = t % 64; int r = t / 64;
    int hin = 2 * h - 1 + r;
    int win = 2 * w0 - 1 + col;
    float v = 0.0f;
    if ((unsigned)hin < 192u && (unsigned)win < 384u)
      v = x[((b * 64 + ci) * 192 + hin) * 384 + win];
    tile[(r * C1_COLS + col) * C1_CIP + ci] = (_Float16)v;
  }
  __syncthreads();

  int wave = tid >> 5;          // m-tile 0..7
  int lane = tid & 31;
  int ln   = lane & 15;
  int kbB  = lane & 16;         // B frag lane-half K offset
  v8f vzero = {0.f, 0.f, 0.f, 0.f, 0.f, 0.f, 0.f, 0.f};
  v8f acc[4] = {vzero, vzero, vzero, vzero};

  for (int kstep = 0; kstep < 18; ++kstep) {  // K = 9 taps * 64ch = 576
    v16h a = *(const v16h*)(packedA + ((kstep * 8 + wave) * 32 + lane) * 16);
    int pos = kstep >> 1;
    int kh = pos / 3, kw = pos % 3;
    int cbase = ((kstep & 1) << 5) + kbB;
    const _Float16* rowp = &tile[(kh * C1_COLS) * C1_CIP];
#pragma unroll
    for (int nt = 0; nt < 4; ++nt) {
      int col = 2 * (nt * 16 + ln) + kw;            // stride-2 conv
      v16h bf = load_bfrag(rowp + col * C1_CIP + cbase);
      acc[nt] = __builtin_amdgcn_wmma_f32_16x16x32_f16(
          false, a, false, bf, (short)0, acc[nt], false, false);
    }
  }

  // Epilogue: * gate1, leaky(0.2), store f16 intermediate.
  int mbase = wave * 16 + ((lane & 16) >> 1);
#pragma unroll
  for (int nt = 0; nt < 4; ++nt) {
    int nn = w0 + nt * 16 + ln;
#pragma unroll
    for (int r = 0; r < 8; ++r) {
      int m = mbase + r;
      float v = acc[nt][r] * g1[b * 128 + m];
      v = v > 0.f ? v : 0.2f * v;
      out2[((b * 128 + m) * 96 + h) * 192 + nn] = (_Float16)v;
    }
  }
}

// ---------------------------------------------------------------------------
// conv2: 3x3 stride1 pad1 128->128 ; plus ident = 1x1 stride2 GEMM (BN folded)
// out6 = leaky( leaky(conv2*g2) + ident + beta )
// ---------------------------------------------------------------------------
#define C2_COLS 66
#define C2_CIP  (128 + LPAD)   // 136
#define DI_CIP  (64 + LPAD)    // 72

__global__ __launch_bounds__(256) void conv2_kernel(
    const _Float16* __restrict__ out2, const float* __restrict__ x,
    const _Float16* __restrict__ packedA2, const _Float16* __restrict__ packedDS,
    const float* __restrict__ g2, const float* __restrict__ beta,
    float* __restrict__ out6) {
  __shared__ _Float16 tile[3 * C2_COLS * C2_CIP];   // conv2 halo
  __shared__ _Float16 itile[64 * DI_CIP];           // ident 1x1 input
  int w0 = blockIdx.x * 64;
  int h  = blockIdx.y;
  int b  = blockIdx.z;
  int tid = threadIdx.x;

  for (int idx = tid; idx < 128 * 3 * C2_COLS; idx += 256) {
    int col = idx % C2_COLS; int t = idx / C2_COLS;
    int r = t % 3; int ci = t / 3;
    int hin = h - 1 + r;
    int win = w0 - 1 + col;
    _Float16 v = (_Float16)0.f;
    if ((unsigned)hin < 96u && (unsigned)win < 192u)
      v = out2[((b * 128 + ci) * 96 + hin) * 192 + win];
    tile[(r * C2_COLS + col) * C2_CIP + ci] = v;
  }
  for (int idx = tid; idx < 64 * 64; idx += 256) {
    int col = idx & 63; int ci = idx >> 6;
    itile[col * DI_CIP + ci] =
        (_Float16)x[((b * 64 + ci) * 192 + 2 * h) * 384 + 2 * (w0 + col)];
  }
  __syncthreads();

  int wave = tid >> 5;
  int lane = tid & 31;
  int ln   = lane & 15;
  int kbB  = lane & 16;
  v8f vzero = {0.f, 0.f, 0.f, 0.f, 0.f, 0.f, 0.f, 0.f};
  v8f acc[4]  = {vzero, vzero, vzero, vzero};
  v8f iacc[4] = {vzero, vzero, vzero, vzero};

  for (int kstep = 0; kstep < 36; ++kstep) {  // K = 9 taps * 128ch = 1152
    v16h a = *(const v16h*)(packedA2 + ((kstep * 8 + wave) * 32 + lane) * 16);
    int pos = kstep >> 2;
    int kh = pos / 3, kw = pos % 3;
    int cbase = ((kstep & 3) << 5) + kbB;
    const _Float16* rowp = &tile[(kh * C2_COLS) * C2_CIP];
#pragma unroll
    for (int nt = 0; nt < 4; ++nt) {
      int col = nt * 16 + ln + kw;
      v16h bf = load_bfrag(rowp + col * C2_CIP + cbase);
      acc[nt] = __builtin_amdgcn_wmma_f32_16x16x32_f16(
          false, a, false, bf, (short)0, acc[nt], false, false);
    }
  }
#pragma unroll
  for (int kstep = 0; kstep < 2; ++kstep) {   // ident: K = 64
    v16h a = *(const v16h*)(packedDS + ((kstep * 8 + wave) * 32 + lane) * 16);
    int cbase = (kstep << 5) + kbB;
#pragma unroll
    for (int nt = 0; nt < 4; ++nt) {
      v16h bf = load_bfrag(itile + (nt * 16 + ln) * DI_CIP + cbase);
      iacc[nt] = __builtin_amdgcn_wmma_f32_16x16x32_f16(
          false, a, false, bf, (short)0, iacc[nt], false, false);
    }
  }

  int mbase = wave * 16 + ((lane & 16) >> 1);
#pragma unroll
  for (int nt = 0; nt < 4; ++nt) {
    int nn = w0 + nt * 16 + ln;
#pragma unroll
    for (int r = 0; r < 8; ++r) {
      int m = mbase + r;
      float v = acc[nt][r] * g2[b * 128 + m];
      v = v > 0.f ? v : 0.2f * v;                 // out4
      v = v + iacc[nt][r] + beta[m];              // + ident (scale folded)
      v = v > 0.f ? v : 0.2f * v;                 // out6
      out6[((b * 128 + m) * 96 + h) * 192 + nn] = v;
    }
  }
}

// ---------------------------------------------------------------------------
extern "C" void kernel_launch(void* const* d_in, const int* in_sizes, int n_in,
                              void* d_out, int out_size, void* d_ws, size_t ws_size,
                              hipStream_t stream) {
  const float* x     = (const float*)d_in[0];
  const float* emb   = (const float*)d_in[1];
  const float* w1    = (const float*)d_in[2];
  const float* w2    = (const float*)d_in[3];
  const float* dsw   = (const float*)d_in[4];
  const float* gamma = (const float*)d_in[5];
  const float* beta  = (const float*)d_in[6];
  const float* g1w   = (const float*)d_in[7];
  const float* g1b   = (const float*)d_in[8];
  const float* g2w   = (const float*)d_in[9];
  const float* g2b   = (const float*)d_in[10];
  float* out = (float*)d_out;
  char* ws = (char*)d_ws;

  _Float16* packed1  = (_Float16*)(ws);                       // 147456 B
  _Float16* packed2  = (_Float16*)(ws + 147456);              // 294912 B
  _Float16* packedDS = (_Float16*)(ws + 147456 + 294912);     // 16384 B
  _Float16* out2     = (_Float16*)(ws + 458752);              // 37748736 B

  float* gates = out + NOUT;   // g1 at [0..1023], g2 at [1024..2047]

  gates_kernel<<<dim3(8, 2), 128, 0, stream>>>(emb, g1w, g1b, g2w, g2b, gates);
  pack_conv3x3<<<(73728 + 255) / 256, 256, 0, stream>>>(w1, packed1, 64, 73728);
  pack_conv3x3<<<(147456 + 255) / 256, 256, 0, stream>>>(w2, packed2, 128, 147456);
  pack_ds<<<32, 256, 0, stream>>>(dsw, gamma, packedDS);

  conv1_kernel<<<dim3(3, 96, 8), 256, 0, stream>>>(x, packed1, gates, out2);
  conv2_kernel<<<dim3(3, 96, 8), 256, 0, stream>>>(out2, x, packed2, packedDS,
                                                   gates + 1024, beta, out);
}
